// POVMProjector_85160611545410
// MI455X (gfx1250) — matrix operand order repaired and checked by reading, not compiled, not run on hardware
//
#include <hip/hip_runtime.h>

typedef __attribute__((ext_vector_type(16))) __bf16 v16bf;
typedef __attribute__((ext_vector_type(8)))  __bf16 v8bf;
typedef __attribute__((ext_vector_type(4)))  __bf16 v4bf;
typedef __attribute__((ext_vector_type(8)))  float  v8f;
typedef __attribute__((ext_vector_type(4)))  float  v4f;

#define NROW   32768      // B*S
#define INDIM  1024
#define OUTDIM 256
#define LDA    264        // padded LDS row stride in bf16 elements
#define PLANE  262144     // 256*1024 elements per bf16 plane in ws

union Frag { v16bf v; v8bf h[2]; };

static __device__ __forceinline__ v8f wmma_bf16(v16bf a, v16bf b, v8f c) {
  return __builtin_amdgcn_wmma_f32_16x16x32_bf16(false, a, false, b, (short)0, c,
                                                 false, false);
}

static __device__ __forceinline__ void splitf(float x, __bf16 &h, __bf16 &l) {
  __bf16 hh = (__bf16)x;
  h = hh;
  l = (__bf16)(x - (float)hh);
}

// ---------------------------------------------------------------------------
// k0: split basis into bf16 hi/lo planes, both (m,i) and transposed (i,m).
// ---------------------------------------------------------------------------
__global__ void __launch_bounds__(256)
k0_basis_split(const float* __restrict__ bre, const float* __restrict__ bim,
               __bf16* __restrict__ ws) {
  int g = blockIdx.x * 256 + threadIdx.x;        // 0 .. 262143  (m*1024 + i)
  int m = g >> 10, i = g & 1023;
  __bf16 rh, rl, ih, il;
  splitf(bre[g], rh, rl);
  splitf(bim[g], ih, il);
  ws[0*PLANE + g] = rh;  ws[1*PLANE + g] = rl;   // (m,i) layout for GEMM1
  ws[2*PLANE + g] = ih;  ws[3*PLANE + g] = il;
  int t = i * 256 + m;                           // (i,m) layout for GEMM2
  ws[4*PLANE + t] = rh;  ws[5*PLANE + t] = rl;
  ws[6*PLANE + t] = ih;  ws[7*PLANE + t] = il;
}

// ---------------------------------------------------------------------------
// k1: inner = psi @ conj(basis)^T, p = |inner|^2, probs = p / rowsum(p).
// One block = 16 rows x all 256 outputs. 8 waves, each 2 N-tiles of 16.
// ---------------------------------------------------------------------------
__global__ void __launch_bounds__(256)
k1_born(const float* __restrict__ pre, const float* __restrict__ pim,
        const __bf16* __restrict__ ws, float* __restrict__ probs) {
  __shared__ __bf16 sReH[16*LDA], sReL[16*LDA], sImH[16*LDA], sImL[16*LDA];
  __shared__ float rowsum[16], rinv[16];

  const int tid = threadIdx.x;
  const int lane = tid & 31, wv = tid >> 5;
  const int lo = lane & 15, hi = lane >> 4;
  const int rowBase = blockIdx.x * 16;
  const int nBase = wv * 32;

  const __bf16 *bReH = ws + 0*PLANE, *bReL = ws + 1*PLANE;
  const __bf16 *bImH = ws + 2*PLANE, *bImL = ws + 3*PLANE;

  if (tid < 16) rowsum[tid] = 0.0f;

  v8f cRe[2] = {}, cP[2] = {}, cM[2] = {};

  for (int kc = 0; kc < INDIM; kc += 256) {
    // stage + split one 16x256 psi chunk (both components)
    #pragma unroll
    for (int e = 0; e < 4; ++e) {
      int f = tid + 256 * e;
      int r = f >> 6, c4 = (f & 63) << 2;
      v4f xr = *(const v4f*)(pre + (size_t)(rowBase + r) * INDIM + kc + c4);
      v4f xi = *(const v4f*)(pim + (size_t)(rowBase + r) * INDIM + kc + c4);
      v4bf rh, rl, ih, il;
      #pragma unroll
      for (int j = 0; j < 4; ++j) {
        __bf16 a, b;
        splitf(xr[j], a, b); rh[j] = a; rl[j] = b;
        splitf(xi[j], a, b); ih[j] = a; il[j] = b;
      }
      int off = r * LDA + c4;
      *(v4bf*)&sReH[off] = rh;  *(v4bf*)&sReL[off] = rl;
      *(v4bf*)&sImH[off] = ih;  *(v4bf*)&sImL[off] = il;
    }
    __syncthreads();

    for (int kk = 0; kk < 256; kk += 32) {
      const int a0 = lo * LDA + kk + 16 * hi;
      const int a1 = a0 + 8;
      Frag aReH, aReL, aImH, aImL;
      aReH.h[0] = *(const v8bf*)&sReH[a0];  aReH.h[1] = *(const v8bf*)&sReH[a1];
      aReL.h[0] = *(const v8bf*)&sReL[a0];  aReL.h[1] = *(const v8bf*)&sReL[a1];
      aImH.h[0] = *(const v8bf*)&sImH[a0];  aImH.h[1] = *(const v8bf*)&sImH[a1];
      aImL.h[0] = *(const v8bf*)&sImL[a0];  aImL.h[1] = *(const v8bf*)&sImL[a1];

      #pragma unroll
      for (int t = 0; t < 2; ++t) {
        size_t be = (size_t)(nBase + 16*t + lo) * INDIM + kc + kk + 16 * hi;
        Frag bRH, bRL, bIH, bIL;
        bRH.h[0] = *(const v8bf*)(bReH + be);  bRH.h[1] = *(const v8bf*)(bReH + be + 8);
        bRL.h[0] = *(const v8bf*)(bReL + be);  bRL.h[1] = *(const v8bf*)(bReL + be + 8);
        bIH.h[0] = *(const v8bf*)(bImH + be);  bIH.h[1] = *(const v8bf*)(bImH + be + 8);
        bIL.h[0] = *(const v8bf*)(bImL + be);  bIL.h[1] = *(const v8bf*)(bImL + be + 8);

        // re  = psi_re*b_re + psi_im*b_im     (split: hh + hl + lh)
        cRe[t] = wmma_bf16(aReH.v, bRH.v, cRe[t]);
        cRe[t] = wmma_bf16(aReH.v, bRL.v, cRe[t]);
        cRe[t] = wmma_bf16(aReL.v, bRH.v, cRe[t]);
        cRe[t] = wmma_bf16(aImH.v, bIH.v, cRe[t]);
        cRe[t] = wmma_bf16(aImH.v, bIL.v, cRe[t]);
        cRe[t] = wmma_bf16(aImL.v, bIH.v, cRe[t]);
        // im+ = psi_im*b_re ; im- = psi_re*b_im ; im = im+ - im-
        cP[t]  = wmma_bf16(aImH.v, bRH.v, cP[t]);
        cP[t]  = wmma_bf16(aImH.v, bRL.v, cP[t]);
        cP[t]  = wmma_bf16(aImL.v, bRH.v, cP[t]);
        cM[t]  = wmma_bf16(aReH.v, bIH.v, cM[t]);
        cM[t]  = wmma_bf16(aReH.v, bIL.v, cM[t]);
        cM[t]  = wmma_bf16(aReL.v, bIH.v, cM[t]);
      }
    }
    __syncthreads();
  }

  // epilogue: p, row sums, normalize, store probs
  float pv[2][8], rs[8];
  #pragma unroll
  for (int v = 0; v < 8; ++v) {
    float re0 = cRe[0][v], im0 = cP[0][v] - cM[0][v];
    float re1 = cRe[1][v], im1 = cP[1][v] - cM[1][v];
    pv[0][v] = re0 * re0 + im0 * im0;
    pv[1][v] = re1 * re1 + im1 * im1;
    float s = pv[0][v] + pv[1][v];
    #pragma unroll
    for (int msk = 1; msk < 16; msk <<= 1) s += __shfl_xor(s, msk, 32);
    rs[v] = s;
  }
  if (lo == 0) {
    #pragma unroll
    for (int v = 0; v < 8; ++v) atomicAdd(&rowsum[v + 8 * hi], rs[v]);
  }
  __syncthreads();
  if (tid < 16) rinv[tid] = 1.0f / (rowsum[tid] + 1e-12f);
  __syncthreads();
  #pragma unroll
  for (int v = 0; v < 8; ++v) {
    float inv = rinv[v + 8 * hi];
    size_t row = rowBase + v + 8 * hi;
    probs[row * OUTDIM + nBase + lo]      = pv[0][v] * inv;
    probs[row * OUTDIM + nBase + 16 + lo] = pv[1][v] * inv;
  }
}

// ---------------------------------------------------------------------------
// k2: collapsed = probs @ basis, then per-row L2 normalize, write interleaved.
// One block = 16 rows x 1024 cols. 16 waves, each 4 N-tiles of 16.
// ---------------------------------------------------------------------------
__global__ void __launch_bounds__(512)
k2_collapse(const float* __restrict__ probs, const __bf16* __restrict__ ws,
            float* __restrict__ out) {
  __shared__ __bf16 sPH[16*LDA], sPL[16*LDA];
  __shared__ float nsum[16], rscale[16];

  const int tid = threadIdx.x;
  const int lane = tid & 31, wv = tid >> 5;
  const int lo = lane & 15, hi = lane >> 4;
  const int rowBase = blockIdx.x * 16;
  const int colBase = wv * 64;

  const __bf16 *tReH = ws + 4*PLANE, *tReL = ws + 5*PLANE;
  const __bf16 *tImH = ws + 6*PLANE, *tImL = ws + 7*PLANE;

  if (tid < 16) nsum[tid] = 0.0f;

  // stage + split probs tile 16x256
  #pragma unroll
  for (int e = 0; e < 2; ++e) {
    int f = tid + 512 * e;
    int r = f >> 6, c4 = (f & 63) << 2;
    v4f x = *(const v4f*)(probs + (size_t)(rowBase + r) * OUTDIM + c4);
    v4bf xh, xl;
    #pragma unroll
    for (int j = 0; j < 4; ++j) { __bf16 a, b; splitf(x[j], a, b); xh[j] = a; xl[j] = b; }
    int off = r * LDA + c4;
    *(v4bf*)&sPH[off] = xh;  *(v4bf*)&sPL[off] = xl;
  }
  __syncthreads();

  v8f cRe[4] = {}, cIm[4] = {};
  for (int kk = 0; kk < OUTDIM; kk += 32) {
    const int a0 = lo * LDA + kk + 16 * hi;
    const int a1 = a0 + 8;
    Frag aH, aL;
    aH.h[0] = *(const v8bf*)&sPH[a0];  aH.h[1] = *(const v8bf*)&sPH[a1];
    aL.h[0] = *(const v8bf*)&sPL[a0];  aL.h[1] = *(const v8bf*)&sPL[a1];

    #pragma unroll
    for (int t = 0; t < 4; ++t) {
      size_t be = (size_t)(colBase + 16*t + lo) * OUTDIM + kk + 16 * hi;
      Frag bRH, bRL, bIH, bIL;
      bRH.h[0] = *(const v8bf*)(tReH + be);  bRH.h[1] = *(const v8bf*)(tReH + be + 8);
      bRL.h[0] = *(const v8bf*)(tReL + be);  bRL.h[1] = *(const v8bf*)(tReL + be + 8);
      bIH.h[0] = *(const v8bf*)(tImH + be);  bIH.h[1] = *(const v8bf*)(tImH + be + 8);
      bIL.h[0] = *(const v8bf*)(tImL + be);  bIL.h[1] = *(const v8bf*)(tImL + be + 8);

      cRe[t] = wmma_bf16(aH.v, bRH.v, cRe[t]);
      cRe[t] = wmma_bf16(aH.v, bRL.v, cRe[t]);
      cRe[t] = wmma_bf16(aL.v, bRH.v, cRe[t]);
      cIm[t] = wmma_bf16(aH.v, bIH.v, cIm[t]);
      cIm[t] = wmma_bf16(aH.v, bIL.v, cIm[t]);
      cIm[t] = wmma_bf16(aL.v, bIH.v, cIm[t]);
    }
  }

  // per-row L2 norm over all 1024 cols
  float rs[8];
  #pragma unroll
  for (int v = 0; v < 8; ++v) {
    float s = 0.0f;
    #pragma unroll
    for (int t = 0; t < 4; ++t) {
      float re = cRe[t][v], im = cIm[t][v];
      s += re * re + im * im;
    }
    #pragma unroll
    for (int msk = 1; msk < 16; msk <<= 1) s += __shfl_xor(s, msk, 32);
    rs[v] = s;
  }
  if (lo == 0) {
    #pragma unroll
    for (int v = 0; v < 8; ++v) atomicAdd(&nsum[v + 8 * hi], rs[v]);
  }
  __syncthreads();
  if (tid < 16) rscale[tid] = 1.0f / (sqrtf(nsum[tid]) + 1e-12f);
  __syncthreads();

  #pragma unroll
  for (int v = 0; v < 8; ++v) {
    float sc = rscale[v + 8 * hi];
    size_t row = rowBase + v + 8 * hi;
    #pragma unroll
    for (int t = 0; t < 4; ++t) {
      int col = colBase + 16 * t + lo;
      float2 o;
      o.x = cRe[t][v] * sc;
      o.y = cIm[t][v] * sc;
      *(float2*)(out + 2 * (row * INDIM + col)) = o;   // interleaved complex64
    }
  }
}

// ---------------------------------------------------------------------------
extern "C" void kernel_launch(void* const* d_in, const int* in_sizes, int n_in,
                              void* d_out, int out_size, void* d_ws, size_t ws_size,
                              hipStream_t stream) {
  const float* psi_re = (const float*)d_in[0];
  const float* psi_im = (const float*)d_in[1];
  const float* bas_re = (const float*)d_in[2];
  const float* bas_im = (const float*)d_in[3];

  float* out   = (float*)d_out;                       // collapsed: 32768*1024*2 floats
  float* probs = out + (size_t)NROW * INDIM * 2;      // probs:     32768*256 floats
  __bf16* ws   = (__bf16*)d_ws;                       // 8 planes * 512KB = 4MB

  k0_basis_split<<<PLANE / 256, 256, 0, stream>>>(bas_re, bas_im, ws);
  k1_born<<<NROW / 16, 256, 0, stream>>>(psi_re, psi_im, ws, probs);
  k2_collapse<<<NROW / 16, 512, 0, stream>>>(probs, ws, out);
}
